// FullNet_68899865362682
// MI455X (gfx1250) — matrix operand order repaired and checked by reading, not compiled, not run on hardware
//
#include <hip/hip_runtime.h>
#include <hip/hip_bf16.h>
#include <stdint.h>

// ---------------------------------------------------------------------------
// Types
// ---------------------------------------------------------------------------
typedef __bf16 v16bf __attribute__((ext_vector_type(16)));
typedef __bf16 v8bf  __attribute__((ext_vector_type(8)));
typedef float  v8f   __attribute__((ext_vector_type(8)));
typedef unsigned int u32;
typedef u32 u32x4 __attribute__((ext_vector_type(4)));
typedef int  i32x4 __attribute__((ext_vector_type(4)));
typedef int  i32x8 __attribute__((ext_vector_type(8)));

#if defined(__has_builtin)
#if __has_builtin(__builtin_amdgcn_tensor_load_to_lds)
#define USE_TDM 1
#else
#define USE_TDM 0
#endif
#else
#define USE_TDM 0
#endif

// ---------------------------------------------------------------------------
// TDM 2-D tile load: global (row-major, 2-byte elems) -> LDS, per ISA D# layout
// group0: count=1 | lds_addr | global_addr[56:32] | type=2
// group1: data_size=2B, tensor_dim0/1, tile_dim0/1, tensor_dim0_stride
// ---------------------------------------------------------------------------
#if USE_TDM
static __device__ __forceinline__ void tdm_load_tile_2d(
    u32 lds_off, const void* gptr, u32 tensor_d0, u32 tensor_d1,
    u32 tile_d0, u32 tile_d1, u32 stride0_elems) {
  unsigned long long ga = (unsigned long long)(uintptr_t)gptr;
  u32x4 g0;
  g0[0] = 1u;                                            // count=1, user desc
  g0[1] = lds_off;                                       // lds_addr (bytes)
  g0[2] = (u32)ga;                                       // global_addr[31:0]
  g0[3] = (u32)((ga >> 32) & 0x01FFFFFFu) | (2u << 30);  // addr[56:32] | type=2
  i32x8 g1;
  g1[0] = (int)(1u << 16);                               // data_size = 2 bytes
  g1[1] = (int)((tensor_d0 & 0xFFFFu) << 16);            // tensor_dim0 lo16
  g1[2] = (int)((tensor_d0 >> 16) | ((tensor_d1 & 0xFFFFu) << 16));
  g1[3] = (int)((tensor_d1 >> 16) | (tile_d0 << 16));    // tile_dim0
  g1[4] = (int)(tile_d1 & 0xFFFFu);                      // tile_dim1 (dim2=0)
  g1[5] = (int)stride0_elems;                            // tensor_dim0_stride
  g1[6] = 0;
  g1[7] = 0;
  i32x4 z4 = {0, 0, 0, 0};
#if __clang_major__ >= 23
  i32x8 z8 = {0, 0, 0, 0, 0, 0, 0, 0};
  __builtin_amdgcn_tensor_load_to_lds(g0, g1, z4, z4, z8, 0);
#else
  __builtin_amdgcn_tensor_load_to_lds(g0, g1, z4, z4, 0);
#endif
}
#endif

// ---------------------------------------------------------------------------
// Kernel 1: sparse input layer.  One block per batch row; LDS histogram of
// T bins; scatter-add w[x-1, u]; + bias; ReLU; emit bf16 row.
// ---------------------------------------------------------------------------
__global__ __launch_bounds__(256) void input_layer_kernel(
    const int* __restrict__ x, const float* __restrict__ w,
    const float* __restrict__ bias, __bf16* __restrict__ h0,
    int U, int T) {
  __shared__ float bins[2048];
  const int b = blockIdx.x;
  for (int t = threadIdx.x; t < T; t += blockDim.x) bins[t] = 0.0f;
  __syncthreads();
  const size_t xrow = (size_t)b * (size_t)U;
  for (int u = threadIdx.x; u < U; u += blockDim.x) {
    int xi = x[xrow + u];
    if (xi > 0) {
      int idx = xi - 1;
      atomicAdd(&bins[idx], w[(size_t)idx * (size_t)U + u]);
    }
  }
  __syncthreads();
  const size_t orow = (size_t)b * (size_t)T;
  for (int t = threadIdx.x; t < T; t += blockDim.x) {
    float v = bins[t] + bias[t];
    h0[orow + t] = (__bf16)fmaxf(v, 0.0f);
  }
}

// ---------------------------------------------------------------------------
// Kernel 2: weight convert + transpose:  Wt[n*K + k] = bf16(W[k*N + n])
// ---------------------------------------------------------------------------
__global__ __launch_bounds__(256) void wconvert_kernel(
    const float* __restrict__ W, __bf16* __restrict__ Wt, int K, int N) {
  int i = blockIdx.x * blockDim.x + threadIdx.x;
  if (i >= K * N) return;
  int k = i / N;
  int n = i - k * N;
  Wt[(size_t)n * (size_t)K + k] = (__bf16)W[i];
}

// ---------------------------------------------------------------------------
// Kernel 3: bf16 WMMA GEMM with bias (+ReLU / bf16 out, or raw f32 out).
//   out[M,N] = act( A[M,K] @ W[K,N] + bias ),  Wt = W^T stored [N,K].
// Block tile 128x64, KT=64, 8 waves; wave (wm,wn) owns a 32x32 patch as four
// v8f accumulators -> 8 v_wmma_f32_16x16x32_bf16 per K-step.
// A/B tiles double-buffered in LDS via the Tensor Data Mover: tile i+1 is
// issued before waiting (s_wait_tensorcnt 2 leaves the 2 in-flight loads of
// tile i+1 outstanding; in-order TENSORcnt completion => tile i is resident).
// ---------------------------------------------------------------------------
template <bool LAST>
__global__ __launch_bounds__(256) void gemm_bf16_kernel(
    const __bf16* __restrict__ A, const __bf16* __restrict__ Wt,
    const float* __restrict__ bias, __bf16* __restrict__ outb,
    float* __restrict__ outf, int M, int N, int K) {
  constexpr int BM = 128, BN = 64, KT = 64;
  constexpr int BUF_ELEMS = (BM + BN) * KT;               // 12288 bf16 = 24 KB
  __shared__ __align__(64) __bf16 smem[2 * BUF_ELEMS];    // 48 KB double buffer

  const int m0 = blockIdx.x * BM;
  const int n0 = blockIdx.y * BN;
  const int wave = threadIdx.x >> 5;
  const int lane = threadIdx.x & 31;
  const int wm = wave & 3;        // 0..3 -> 32-row slice
  const int wn = wave >> 2;       // 0..1 -> 32-col slice
  const int half = lane >> 4;     // K-half selector per WMMA layout
  const int l16 = lane & 15;

  v8f acc[2][2] = {{v8f{}, v8f{}}, {v8f{}, v8f{}}};

  const int ntiles = K / KT;

#if USE_TDM
  auto issue = [&](int buf, int k0) {
    u32 base = (u32)(buf * BUF_ELEMS * 2);
    tdm_load_tile_2d(base, A + (size_t)m0 * K + k0, (u32)K, (u32)M,
                     (u32)KT, (u32)BM, (u32)K);
    tdm_load_tile_2d(base + (u32)(BM * KT * 2), Wt + (size_t)n0 * K + k0,
                     (u32)K, (u32)N, (u32)KT, (u32)BN, (u32)K);
  };
  if (threadIdx.x < 32) issue(0, 0);   // prologue: tile 0 -> buffer 0
#endif

  for (int i = 0; i < ntiles; ++i) {
    const int buf = i & 1;
    __syncthreads();  // everyone done reading buf^1 (tile i-1): safe to refill
#if USE_TDM
    if (threadIdx.x < 32) {
      const bool more = (i + 1) < ntiles;
      if (more) issue(buf ^ 1, (i + 1) * KT);
      // Tile i resident when <=2 tensor ops (tile i+1's) remain outstanding.
      if (more) {
        __builtin_amdgcn_s_wait_tensorcnt(2);
        __builtin_prefetch(A + (size_t)m0 * K + (i + 1) * KT, 0, 3);
        __builtin_prefetch(Wt + (size_t)n0 * K + (i + 1) * KT, 0, 3);
      } else {
        __builtin_amdgcn_s_wait_tensorcnt(0);
      }
    }
#else
    {
      __bf16* Alds_w = smem + buf * BUF_ELEMS;
      __bf16* Blds_w = Alds_w + BM * KT;
      const __bf16* Ag = A + (size_t)m0 * K + i * KT;
      for (int t = threadIdx.x; t < (BM * KT) / 8; t += 256) {
        int r = t / (KT / 8), c = t % (KT / 8);
        *(v8bf*)&Alds_w[r * KT + c * 8] =
            *(const v8bf*)&Ag[(size_t)r * K + c * 8];
      }
      const __bf16* Bg = Wt + (size_t)n0 * K + i * KT;
      for (int t = threadIdx.x; t < (BN * KT) / 8; t += 256) {
        int r = t / (KT / 8), c = t % (KT / 8);
        *(v8bf*)&Blds_w[r * KT + c * 8] =
            *(const v8bf*)&Bg[(size_t)r * K + c * 8];
      }
    }
#endif
    __syncthreads();  // tile i visible to all waves

    const __bf16* Alds = smem + buf * BUF_ELEMS;
    const __bf16* Blds = Alds + BM * KT;

#pragma unroll
    for (int kk = 0; kk < KT; kk += 32) {
      // B fragments: lane holds column n, 16 contiguous K values.
      v16bf bf[2];
#pragma unroll
      for (int ns = 0; ns < 2; ++ns) {
        bf[ns] = *(const v16bf*)
            &Blds[(wn * 32 + ns * 16 + l16) * KT + kk + half * 16];
      }
#pragma unroll
      for (int ms = 0; ms < 2; ++ms) {
        // A fragment: elems 0..15 = K half*16..+15 (matches B), row m = l16.
        int row = wm * 32 + ms * 16 + l16;
        v8bf lo = *(const v8bf*)&Alds[row * KT + kk + half * 16];
        v8bf hi = *(const v8bf*)&Alds[row * KT + kk + half * 16 + 8];
        v16bf af;
#pragma unroll
        for (int e = 0; e < 8; ++e) {
          af[e] = lo[e];
          af[e + 8] = hi[e];
        }
#pragma unroll
        for (int ns = 0; ns < 2; ++ns) {
          acc[ms][ns] = __builtin_amdgcn_wmma_f32_16x16x32_bf16(
              false, af, false, bf[ns], (short)0, acc[ms][ns], false, false);
        }
      }
    }
  }

  // Epilogue: C/D layout -> lane l16 = n, VGPR r -> m = half*8 + r.
#pragma unroll
  for (int ns = 0; ns < 2; ++ns) {
    const int gn = n0 + wn * 32 + ns * 16 + l16;
    const float bv = bias[gn];
#pragma unroll
    for (int ms = 0; ms < 2; ++ms) {
#pragma unroll
      for (int r = 0; r < 8; ++r) {
        int gm = m0 + wm * 32 + ms * 16 + half * 8 + r;
        float v = acc[ms][ns][r] + bv;
        if (LAST) {
          outf[(size_t)gm * N + gn] = v;
        } else {
          outb[(size_t)gm * N + gn] = (__bf16)fmaxf(v, 0.0f);
        }
      }
    }
  }
}

// ---------------------------------------------------------------------------
// Host launch
// ---------------------------------------------------------------------------
extern "C" void kernel_launch(void* const* d_in, const int* in_sizes, int n_in,
                              void* d_out, int out_size, void* d_ws,
                              size_t ws_size, hipStream_t stream) {
  (void)in_sizes; (void)n_in; (void)out_size; (void)ws_size;
  const int B = 8192, U = 4096, T = 2048;
  const int M1 = 512, M2 = 512, M3 = 512, M4 = 256, M5 = 256;

  const int*   x  = (const int*)d_in[0];
  const float* w  = (const float*)d_in[1];
  const float* b1 = (const float*)d_in[2];
  const float* W2 = (const float*)d_in[3];  const float* b2 = (const float*)d_in[4];
  const float* W3 = (const float*)d_in[5];  const float* b3 = (const float*)d_in[6];
  const float* W4 = (const float*)d_in[7];  const float* b4 = (const float*)d_in[8];
  const float* W5 = (const float*)d_in[9];  const float* b5 = (const float*)d_in[10];
  const float* W6 = (const float*)d_in[11]; const float* b6 = (const float*)d_in[12];
  const float* W7 = (const float*)d_in[13]; const float* b7 = (const float*)d_in[14];
  float* out = (float*)d_out;

  // Workspace layout (~21.5 MB): bf16 transposed weights + two ping-pong
  // activation buffers.  H0 (bf16, 32 MB) lives in the front of d_out (64 MB
  // fp32 buffer) — dead before the final layer overwrites d_out.
  char* ws = (char*)d_ws;
  size_t off = 0;
  auto take = [&](size_t bytes) {
    size_t p = off;
    off = (off + bytes + 255) & ~(size_t)255;
    return p;
  };
  __bf16* W2t = (__bf16*)(ws + take((size_t)T * M1 * 2));
  __bf16* W3t = (__bf16*)(ws + take((size_t)M1 * M2 * 2));
  __bf16* W4t = (__bf16*)(ws + take((size_t)M2 * M3 * 2));
  __bf16* W5t = (__bf16*)(ws + take((size_t)M3 * M4 * 2));
  __bf16* W6t = (__bf16*)(ws + take((size_t)M4 * M5 * 2));
  __bf16* W7t = (__bf16*)(ws + take((size_t)M5 * T * 2));
  __bf16* buf1 = (__bf16*)(ws + take((size_t)B * M1 * 2));  // H1/H3/H5
  __bf16* buf2 = (__bf16*)(ws + take((size_t)B * M1 * 2));  // H2/H4
  __bf16* H0 = (__bf16*)d_out;

  // 1) sparse input layer -> H0 (bf16, relu'd)
  input_layer_kernel<<<B, 256, 0, stream>>>(x, w, b1, H0, U, T);

  // 2) weight convert + transpose
  auto cvt = [&](const float* W, __bf16* Wt, int K, int N) {
    int n = K * N;
    wconvert_kernel<<<(n + 255) / 256, 256, 0, stream>>>(W, Wt, K, N);
  };
  cvt(W2, W2t, T, M1);
  cvt(W3, W3t, M1, M2);
  cvt(W4, W4t, M2, M3);
  cvt(W5, W5t, M3, M4);
  cvt(W6, W6t, M4, M5);
  cvt(W7, W7t, M5, T);

  // 3) GEMM chain (block tile 128x64, KT=64, double-buffered TDM)
  auto gemm = [&](const __bf16* Ain, const __bf16* Wt, const float* bias,
                  __bf16* ob, int M, int N, int K) {
    dim3 grid(M / 128, N / 64);
    gemm_bf16_kernel<false><<<grid, 256, 0, stream>>>(Ain, Wt, bias, ob,
                                                      nullptr, M, N, K);
  };
  gemm(H0,   W2t, b2, buf1, B, M1, T);    // H1
  gemm(buf1, W3t, b3, buf2, B, M2, M1);   // H2
  gemm(buf2, W4t, b4, buf1, B, M3, M2);   // H3
  gemm(buf1, W5t, b5, buf2, B, M4, M3);   // H4
  gemm(buf2, W6t, b6, buf1, B, M5, M4);   // H5
  {
    dim3 grid(B / 128, T / 64);
    gemm_bf16_kernel<true><<<grid, 256, 0, stream>>>(buf1, W7t, b7, nullptr,
                                                     out, B, T, M5);
  }
}
